// Mamba2Block_1726576854845
// MI455X (gfx1250) — matrix operand order repaired and checked by reading, not compiled, not run on hardware
//
#include <hip/hip_runtime.h>
#include <math.h>

#define D_MODEL   2048
#define D_INNER   4096
#define NH        64
#define HP        64
#define DS        128
#define DCONV     4
#define CHUNK     256
#define CONV_DIM  (D_INNER + 2*DS)          // 4352
#define D_IN_PROJ (2*D_INNER + 2*DS + NH)   // 8512
#define BSZ       2
#define SEQ       2048
#define NC        (SEQ/CHUNK)               // 8
#define NTOK      (BSZ*SEQ)                 // 4096

typedef __attribute__((ext_vector_type(16))) __bf16 v16bf;
typedef __attribute__((ext_vector_type(8)))  float  v8f;

union FragU { unsigned short u[16]; uint4 q[2]; v16bf v; };

__device__ __forceinline__ unsigned short f2b(float f) {
    unsigned int u = __float_as_uint(f);
    unsigned int r = u + 0x7FFFu + ((u >> 16) & 1u);
    return (unsigned short)(r >> 16);
}
__device__ __forceinline__ float b2f(unsigned short h) {
    return __uint_as_float(((unsigned int)h) << 16);
}

// Swizzled LDS tiles: one 16(m)x32(k) A-tile or 32(k)x16(n) B-tile = 512 halfs.
// Lane l's whole WMMA fragment is tile[l*16 .. l*16+15] -> 2x ds_load_b128.
// A: lane=(m&15)|(k&8?16:0), pos=(k&7)|(k&16?8:0)
// B: lane=(n&15)|(k&8?16:0), pos=(k&7)|(k&16?8:0)
__device__ __forceinline__ v16bf load_frag(const unsigned short* tile) {
    int lane = threadIdx.x & 31;
    FragU f;
    f.q[0] = *(const uint4*)(tile + lane * 16);
    f.q[1] = *(const uint4*)(tile + lane * 16 + 8);
    return f.v;
}

__device__ __forceinline__ v8f wmma_bf16(v16bf a, v16bf b, v8f c) {
    return __builtin_amdgcn_wmma_f32_16x16x32_bf16(false, a, false, b, (short)0, c,
                                                   false, false);
}

// ---------------- f32 -> bf16 convert ----------------
__global__ void f2b_convert_kernel(const float* __restrict__ in,
                                   unsigned short* __restrict__ out, size_t n) {
    size_t i = (size_t)blockIdx.x * blockDim.x + threadIdx.x;
    size_t stride = (size_t)gridDim.x * blockDim.x;
    for (; i < n; i += stride) out[i] = f2b(in[i]);
}

// ------- GEMM: C[M,N] = A[M,K] @ W[N,K]^T (bf16 in, f32 acc)
// Block tile 128x64, 8 waves each computing 2x2 16x16 tiles (4 WMMA / K-step),
// double-buffered swizzled LDS + prefetch.
template <bool OUT_F32>
__global__ __launch_bounds__(256)
void gemm_xwt_kernel(const unsigned short* __restrict__ A,
                     const unsigned short* __restrict__ W,
                     void* __restrict__ Cout, int M, int N, int K) {
    __shared__ __align__(16) unsigned short Asw[2*8*512];   // 2 bufs x 8 m-tiles
    __shared__ __align__(16) unsigned short Bsw[2*4*512];   // 2 bufs x 4 n-tiles
    int n0 = blockIdx.x * 64;
    int m0 = blockIdx.y * 128;
    int tid  = threadIdx.x;
    int wave = tid >> 5;
    int mtb  = (wave & 3) * 2;     // 2 m-tiles per wave
    int ntb  = (wave >> 2) * 2;    // 2 n-tiles per wave
    // A staging: 128 rows x 32 k, thread t -> row=t>>1, ka=(t&1)*16 (2x b128)
    int arow = tid >> 1;
    int ka   = (tid & 1) * 16;
    int a_dst = (arow >> 4)*512 + (arow & 15)*16 + ((ka & 16) ? 8 : 0);
    const unsigned short* aptr = A + (size_t)(m0 + arow) * K + ka;
    // B staging: 64 rows x 32 k, thread t -> nn=t>>2, kb=(t&3)*8 (1x b128)
    int nn = tid >> 2;
    int kb = (tid & 3) * 8;
    int b_dst = (nn >> 4)*512 + (((nn & 15) | ((kb & 8) ? 16 : 0)) * 16) + ((kb & 16) ? 8 : 0);
    const unsigned short* wptr = W + (size_t)(n0 + nn) * K + kb;
    v8f acc[2][2] = {};
    {
        uint4 q0 = *(const uint4*)aptr;
        uint4 q1 = *(const uint4*)(aptr + 8);
        *(uint4*)&Asw[a_dst]       = q0;   // lane-lo run
        *(uint4*)&Asw[a_dst + 256] = q1;   // lane-hi run (+16 lanes * 16 halfs)
        *(uint4*)&Bsw[b_dst]       = *(const uint4*)wptr;
    }
    __syncthreads();
    int cur = 0;
    for (int k0 = 0; k0 < K; k0 += 32) {
        if (k0 + 32 < K) {
            __builtin_prefetch(aptr + k0 + 96, 0, 1);
            __builtin_prefetch(wptr + k0 + 96, 0, 1);
            uint4 q0 = *(const uint4*)(aptr + k0 + 32);
            uint4 q1 = *(const uint4*)(aptr + k0 + 40);
            int ab = (1 - cur) * 4096;
            *(uint4*)&Asw[ab + a_dst]       = q0;
            *(uint4*)&Asw[ab + a_dst + 256] = q1;
            *(uint4*)&Bsw[(1 - cur)*2048 + b_dst] = *(const uint4*)(wptr + k0 + 32);
        }
        v16bf a0 = load_frag(&Asw[cur*4096 + mtb*512]);
        v16bf a1 = load_frag(&Asw[cur*4096 + (mtb + 1)*512]);
        v16bf b0 = load_frag(&Bsw[cur*2048 + ntb*512]);
        v16bf b1 = load_frag(&Bsw[cur*2048 + (ntb + 1)*512]);
        acc[0][0] = wmma_bf16(a0, b0, acc[0][0]);
        acc[0][1] = wmma_bf16(a0, b1, acc[0][1]);
        acc[1][0] = wmma_bf16(a1, b0, acc[1][0]);
        acc[1][1] = wmma_bf16(a1, b1, acc[1][1]);
        __syncthreads();
        cur ^= 1;
    }
    int lane = tid & 31;
    int nloc = lane & 15;
    int mofs = (lane < 16) ? 0 : 8;
#pragma unroll
    for (int mi = 0; mi < 2; mi++)
#pragma unroll
        for (int r = 0; r < 8; r++) {
            int m  = m0 + (mtb + mi)*16 + r + mofs;
            int na = n0 + ntb*16 + nloc;
            if (OUT_F32) {
                float* C = (float*)Cout;
                C[(size_t)m*N + na]      = acc[mi][0][r];
                C[(size_t)m*N + na + 16] = acc[mi][1][r];
            } else {
                unsigned short* C = (unsigned short*)Cout;
                C[(size_t)m*N + na]      = f2b(acc[mi][0][r]);
                C[(size_t)m*N + na + 16] = f2b(acc[mi][1][r]);
            }
        }
}

// ---------------- depthwise conv4 + SiLU ----------------
__global__ void conv_silu_kernel(const unsigned short* __restrict__ zx,
                                 const float* __restrict__ cw,
                                 const float* __restrict__ cb,
                                 unsigned short* __restrict__ xbc) {
    int idx = blockIdx.x * blockDim.x + threadIdx.x;
    if (idx >= NTOK * CONV_DIM) return;
    int c = idx % CONV_DIM;
    int t = idx / CONV_DIM;
    int b = t / SEQ;
    int l = t % SEQ;
    float acc = cb[c];
#pragma unroll
    for (int k = 0; k < DCONV; k++) {
        int ll = l - (DCONV - 1) + k;
        if (ll >= 0) {
            float xv = b2f(zx[(size_t)(b*SEQ + ll)*D_IN_PROJ + D_INNER + c]);
            acc += xv * cw[c*DCONV + k];
        }
    }
    float s = acc / (1.0f + __expf(-acc));
    xbc[(size_t)t*CONV_DIM + c] = f2b(s);
}

// ---------------- dt = softplus, dA cumulative sum per chunk ----------------
__global__ __launch_bounds__(256)
void dt_cumsum_kernel(const unsigned short* __restrict__ zx,
                      const float* __restrict__ dt_bias,
                      const float* __restrict__ A_log,
                      float* __restrict__ dtc, float* __restrict__ dAcs,
                      float* __restrict__ cdecay) {
    int blk = blockIdx.x;                 // (b*NC+c)*NH + h
    int h = blk % NH;
    int c = (blk / NH) % NC;
    int b = blk / (NH * NC);
    int t = threadIdx.x;
    __shared__ float sbuf[2][CHUNK];
    int tok = b*SEQ + c*CHUNK + t;
    float raw = b2f(zx[(size_t)tok*D_IN_PROJ + (D_IN_PROJ - NH) + h]) + dt_bias[h];
    float dtv = (raw > 20.0f) ? raw : log1pf(__expf(raw));
    float dA  = dtv * (-__expf(A_log[h]));
    sbuf[0][t] = dA;
    __syncthreads();
    int src = 0;
    for (int off = 1; off < CHUNK; off <<= 1) {
        float v = sbuf[src][t];
        if (t >= off) v += sbuf[src][t - off];
        sbuf[1 - src][t] = v;
        src = 1 - src;
        __syncthreads();
    }
    float cs = sbuf[src][t];
    size_t base = (size_t)blk * CHUNK;
    dtc[base + t]  = dtv;
    dAcs[base + t] = cs;
    if (t == CHUNK - 1) cdecay[blk] = __expf(cs);
}

// ---------------- states[hp,n] = sum_j w[j]*x[j,hp]*B[j,n] (WMMA) ----------
__global__ __launch_bounds__(256)
void states_kernel(const unsigned short* __restrict__ xbc,
                   const float* __restrict__ dtc, const float* __restrict__ dAcs,
                   float* __restrict__ states) {
    int blk = blockIdx.x;                 // (b*NC+c)*NH + h
    int h  = blk % NH;
    int bc = blk / NH;
    int b  = bc / NC;
    int c  = bc % NC;
    __shared__ __align__(16) unsigned short Asw[4*512];   // x^T, 4 m-tiles
    __shared__ __align__(16) unsigned short Bsw[8*512];   // w*B, 8 n-tiles
    __shared__ float wbuf[CHUNK];
    int tid = threadIdx.x;
    {
        size_t base = (size_t)blk * CHUNK;
        float last = dAcs[base + CHUNK - 1];
        wbuf[tid] = __expf(last - dAcs[base + tid]) * dtc[base + tid];
    }
    __syncthreads();
    int wave = tid >> 5;
    int mt   = wave & 3;
    int ntb  = (wave >> 2) * 4;
    v8f acc[4] = {{}, {}, {}, {}};
    size_t tokbase = (size_t)(b*SEQ + c*CHUNK);
    int kk  = tid >> 3;                   // 0..31
    int qi  = tid & 7;
    for (int k0 = 0; k0 < CHUNK; k0 += 32) {
        {   // A swizzle: element (m=hp, k=kk), transpose of x row
            int hp0 = qi * 8;
            __align__(16) unsigned short tmp[8];
            *(uint4*)tmp = *(const uint4*)(xbc + (tokbase + k0 + kk)*CONV_DIM + h*HP + hp0);
            int posA   = (kk & 7) | ((kk & 16) ? 8 : 0);
            int laneHi = (kk & 8) ? 16 : 0;
#pragma unroll
            for (int i = 0; i < 8; i++) {
                int hp = hp0 + i;
                Asw[(hp >> 4)*512 + ((hp & 15) | laneHi)*16 + posA] = tmp[i];
            }
        }
        {   // B swizzle with w multiplied
            int n0l = qi * 16;
            const unsigned short* src = xbc + (tokbase + k0 + kk)*CONV_DIM + D_INNER + n0l;
            __align__(16) unsigned short tmp[16];
            *(uint4*)&tmp[0] = *(const uint4*)src;
            *(uint4*)&tmp[8] = *(const uint4*)(src + 8);
            float w = wbuf[k0 + kk];
            int posB   = (kk & 7) | ((kk & 16) ? 8 : 0);
            int laneHi = (kk & 8) ? 16 : 0;
            int tb     = (n0l >> 4) * 512;
#pragma unroll
            for (int i = 0; i < 16; i++)
                Bsw[tb + (i | laneHi)*16 + posB] = f2b(b2f(tmp[i]) * w);
        }
        __syncthreads();
        v16bf a = load_frag(&Asw[mt*512]);
#pragma unroll
        for (int j = 0; j < 4; j++) {
            v16bf bb = load_frag(&Bsw[(ntb + j)*512]);
            acc[j] = wmma_bf16(a, bb, acc[j]);
        }
        __syncthreads();
    }
    int lane = tid & 31;
    int nloc = lane & 15;
    int mofs = (lane < 16) ? 0 : 8;
    float* out = states + (size_t)blk * HP * DS;
#pragma unroll
    for (int j = 0; j < 4; j++)
#pragma unroll
        for (int r = 0; r < 8; r++) {
            int hp = mt*16 + r + mofs;
            int n  = (ntb + j)*16 + nloc;
            out[hp*DS + n] = acc[j][r];
        }
}

// ---------------- inter-chunk state scan (8 sequential steps) ---------------
__global__ __launch_bounds__(256)
void hscan_kernel(const float* __restrict__ states, const float* __restrict__ cdecay,
                  float* __restrict__ hprev) {
    int bh = blockIdx.x;                  // b*NH + h
    int b = bh / NH;
    int h = bh % NH;
    for (int e = threadIdx.x; e < HP*DS; e += 256) {
        float hst = 0.0f;
        for (int c = 0; c < NC; c++) {
            size_t blk = ((size_t)(b*NC + c))*NH + h;
            hprev[blk*HP*DS + e] = hst;
            hst = hst * cdecay[blk] + states[blk*HP*DS + e];
        }
    }
}

// ------- fused: scores = (C@B^T)*mask, y = scores@x + exp(dAcs)*(C@h^T) + D*x
__global__ __launch_bounds__(256)
void y_chunk_kernel(const unsigned short* __restrict__ xbc,
                    const float* __restrict__ dtc, const float* __restrict__ dAcs,
                    const float* __restrict__ hprev, const float* __restrict__ Dp,
                    unsigned short* __restrict__ y) {
    int blk = blockIdx.x;
    int ib  = blk & 3;                    // 64-row i-block within chunk
    int bch = blk >> 2;                   // (b*NC+c)*NH + h
    int h  = bch % NH;
    int bc = bch / NH;
    int b  = bc / NC;
    int c  = bc % NC;
    int i0 = ib * 64;
    // A-swizzled: Csw = 4 m-tiles x 4 k-tiles; Ssw = 4 m-tiles x 8 k-tiles
    __shared__ __align__(16) unsigned short Csw[16*512];
    __shared__ __align__(16) unsigned short Ssw[32*512];
    __shared__ __align__(16) unsigned short Tsw[8*512];   // B-swizzled staging
    __shared__ float dABuf[CHUNK];
    __shared__ float dtBuf[CHUNK];
    int tid  = threadIdx.x;
    int wave = tid >> 5;
    int lane = tid & 31;
    size_t tokbase = (size_t)(b*SEQ + c*CHUNK);
    size_t dbase   = (size_t)bch * CHUNK;
    dABuf[tid] = dAcs[dbase + tid];
    dtBuf[tid] = dtc[dbase + tid];
    {   // stage C block (A-swizzle): 4x b128 stores per thread
        int rr = tid >> 2;                // m 0..63
        int nn = (tid & 3) * 32;          // k base
        const unsigned short* src = xbc + (tokbase + i0 + rr)*CONV_DIM + D_INNER + DS + nn;
        int mtile = rr >> 4, ml = rr & 15, kt = nn >> 5;
#pragma unroll
        for (int r8 = 0; r8 < 4; r8++) {
            uint4 d = *(const uint4*)(src + r8*8);
            int k = r8 * 8;               // local k within 32-k tile
            int laneHi = (k & 8) ? 16 : 0;
            int pos    = (k & 16) ? 8 : 0;
            *(uint4*)&Csw[(mtile*4 + kt)*512 + ((ml | laneHi)*16) + pos] = d;
        }
    }
    __syncthreads();

    // Phase 1: masked scores into Ssw, processed in two 128-col halves
    {
        int mt  = wave & 3;
        int ntq = (wave >> 2) * 4;
        for (int half = 0; half < 2; half++) {
            v8f sacc[4] = {{}, {}, {}, {}};
            for (int k0 = 0; k0 < DS; k0 += 32) {
                {   // Tsw[j-tile] (B-swizzle): 2x b128 stores per thread
                    int jl = tid >> 1;
                    int kb = (tid & 1) * 16;
                    const unsigned short* src =
                        xbc + (tokbase + half*128 + jl)*CONV_DIM + D_INNER + k0 + kb;
                    int lsw = (jl & 15) * 16 + (kb ? 8 : 0);
                    int tb  = (jl >> 4) * 512;
                    *(uint4*)&Tsw[tb + lsw]       = *(const uint4*)src;
                    *(uint4*)&Tsw[tb + lsw + 256] = *(const uint4*)(src + 8);
                }
                __syncthreads();
                v16bf a = load_frag(&Csw[(mt*4 + (k0 >> 5))*512]);
#pragma unroll
                for (int j = 0; j < 4; j++) {
                    v16bf bb = load_frag(&Tsw[(ntq + j)*512]);
                    sacc[j] = wmma_bf16(a, bb, sacc[j]);
                }
                __syncthreads();
            }
            int nloc = lane & 15;
            int mofs = (lane < 16) ? 0 : 8;
#pragma unroll
            for (int j = 0; j < 4; j++)
#pragma unroll
                for (int r = 0; r < 8; r++) {
                    int il = mt*16 + r + mofs;
                    int ig = i0 + il;
                    int jj = half*128 + (ntq + j)*16 + nloc;
                    float f = (jj <= ig) ? __expf(dABuf[ig] - dABuf[jj]) * dtBuf[jj] : 0.0f;
                    int tile = (il >> 4)*8 + (jj >> 5);
                    Ssw[tile*512 + ((il & 15) | ((jj & 8) ? 16 : 0))*16
                        + ((jj & 7) | ((jj & 16) ? 8 : 0))] = f2b(sacc[j][r] * f);
                }
        }
    }
    __syncthreads();

    int mt   = wave & 3;
    int ntb2 = (wave >> 2) * 2;
    v8f y1a = {}; v8f y1b = {};
    // Phase 2: y1 = Ssw(64x256) @ x(256x64)
    for (int k0 = 0; k0 < CHUNK; k0 += 32) {
        {   // stage x tile (B-swizzle, genuine transpose -> scalar stores)
            int kk = tid >> 3;
            int p0 = (tid & 7) * 8;
            __align__(16) unsigned short tmp[8];
            *(uint4*)tmp = *(const uint4*)(xbc + (tokbase + k0 + kk)*CONV_DIM + h*HP + p0);
            int posX   = (kk & 7) | ((kk & 16) ? 8 : 0);
            int laneHi = (kk & 8) ? 16 : 0;
#pragma unroll
            for (int i = 0; i < 8; i++) {
                int p = p0 + i;
                Tsw[(p >> 4)*512 + ((p & 15) | laneHi)*16 + posX] = tmp[i];
            }
        }
        __syncthreads();
        v16bf a  = load_frag(&Ssw[(mt*8 + (k0 >> 5))*512]);
        v16bf b0 = load_frag(&Tsw[ntb2*512]);
        v16bf b1 = load_frag(&Tsw[(ntb2 + 1)*512]);
        y1a = wmma_bf16(a, b0, y1a);
        y1b = wmma_bf16(a, b1, y1b);
        __syncthreads();
    }
    // Phase 3: U = C(64x128) @ hprev^T(128x64)
    v8f ua = {}; v8f ub = {};
    const float* hbase = hprev + (size_t)bch * HP * DS;
    for (int k0 = 0; k0 < DS; k0 += 32) {
        {   // stage h^T tile (B-swizzle): f32->bf16, one b128 store per thread
            int p   = tid >> 2;
            int kn0 = (tid & 3) * 8;
            const float* src = hbase + (size_t)p*DS + k0 + kn0;
            __align__(16) unsigned short tmp[8];
#pragma unroll
            for (int i = 0; i < 8; i++) tmp[i] = f2b(src[i]);
            int lsw = ((p & 15) | ((kn0 & 8) ? 16 : 0)) * 16 + ((kn0 & 16) ? 8 : 0);
            *(uint4*)&Tsw[(p >> 4)*512 + lsw] = *(uint4*)tmp;
        }
        __syncthreads();
        v16bf a  = load_frag(&Csw[(mt*4 + (k0 >> 5))*512]);
        v16bf b0 = load_frag(&Tsw[ntb2*512]);
        v16bf b1 = load_frag(&Tsw[(ntb2 + 1)*512]);
        ua = wmma_bf16(a, b0, ua);
        ub = wmma_bf16(a, b1, ub);
        __syncthreads();
    }
    // Epilogue: y = y1 + exp(dAcs[i])*U + Dp*x
    {
        float dp = Dp[h];
        int nloc = lane & 15;
        int mofs = (lane < 16) ? 0 : 8;
#pragma unroll
        for (int r = 0; r < 8; r++) {
            int il = mt*16 + r + mofs;
            int ig = i0 + il;
            float ef = __expf(dABuf[ig]);
            int p0 = ntb2*16 + nloc;
            int p1 = p0 + 16;
            float x0 = b2f(xbc[(tokbase + ig)*CONV_DIM + h*HP + p0]);
            float x1 = b2f(xbc[(tokbase + ig)*CONV_DIM + h*HP + p1]);
            y[(tokbase + ig)*D_INNER + h*HP + p0] = f2b(y1a[r] + ef*ua[r] + dp*x0);
            y[(tokbase + ig)*D_INNER + h*HP + p1] = f2b(y1b[r] + ef*ub[r] + dp*x1);
        }
    }
}

// ---------------- gating (y * silu(z)) + RMS norm ----------------
__global__ __launch_bounds__(256)
void gatenorm_kernel(const unsigned short* __restrict__ zx,
                     const float* __restrict__ norm_w,
                     unsigned short* __restrict__ y) {
    int row = blockIdx.x;
    int tid = threadIdx.x;
    __shared__ float red[8];
    float vals[16];
    float ss = 0.0f;
#pragma unroll
    for (int i = 0; i < 16; i++) {
        int col = tid + i*256;
        float yv = b2f(y[(size_t)row*D_INNER + col]);
        float zv = b2f(zx[(size_t)row*D_IN_PROJ + col]);
        float g = zv / (1.0f + __expf(-zv));
        float v = yv * g;
        vals[i] = v;
        ss += v*v;
    }
#pragma unroll
    for (int off = 16; off > 0; off >>= 1) ss += __shfl_down(ss, off, 32);
    if ((tid & 31) == 0) red[tid >> 5] = ss;
    __syncthreads();
    if (tid == 0) {
        float s = 0.0f;
        for (int i = 0; i < 8; i++) s += red[i];
        red[0] = s;
    }
    __syncthreads();
    float scale = rsqrtf(red[0] / (float)D_INNER + 1e-5f);
#pragma unroll
    for (int i = 0; i < 16; i++) {
        int col = tid + i*256;
        y[(size_t)row*D_INNER + col] = f2b(vals[i] * scale * norm_w[col]);
    }
}

// ---------------- workspace layout ----------------
static const size_t OFF_U16    = 0;
static const size_t OFF_WIN16  = OFF_U16    + (size_t)NTOK*D_MODEL*2;
static const size_t OFF_WOUT16 = OFF_WIN16  + (size_t)D_IN_PROJ*D_MODEL*2;
static const size_t OFF_ZX16   = OFF_WOUT16 + (size_t)D_MODEL*D_INNER*2;
static const size_t OFF_XBC16  = OFF_ZX16   + (size_t)NTOK*D_IN_PROJ*2;
static const size_t OFF_DTC    = OFF_XBC16  + (size_t)NTOK*CONV_DIM*2;
static const size_t OFF_DACS   = OFF_DTC    + (size_t)BSZ*NC*NH*CHUNK*4;
static const size_t OFF_CDEC   = OFF_DACS   + (size_t)BSZ*NC*NH*CHUNK*4;
static const size_t OFF_STATES = OFF_CDEC   + (size_t)BSZ*NC*NH*4 + 3840; // pad; 256-B aligned
static const size_t OFF_HPREV  = OFF_STATES + (size_t)BSZ*NC*NH*HP*DS*4;
static const size_t OFF_Y16    = OFF_HPREV  + (size_t)BSZ*NC*NH*HP*DS*4;

extern "C" void kernel_launch(void* const* d_in, const int* in_sizes, int n_in,
                              void* d_out, int out_size, void* d_ws, size_t ws_size,
                              hipStream_t stream) {
    (void)in_sizes; (void)n_in; (void)out_size; (void)ws_size;
    const float* u       = (const float*)d_in[0];
    const float* W_in    = (const float*)d_in[1];
    const float* conv_w  = (const float*)d_in[2];
    const float* conv_b  = (const float*)d_in[3];
    const float* dt_bias = (const float*)d_in[4];
    const float* A_log   = (const float*)d_in[5];
    const float* Dp      = (const float*)d_in[6];
    const float* norm_w  = (const float*)d_in[7];
    const float* W_out   = (const float*)d_in[8];

    char* ws = (char*)d_ws;
    unsigned short* u16    = (unsigned short*)(ws + OFF_U16);
    unsigned short* Win16  = (unsigned short*)(ws + OFF_WIN16);
    unsigned short* Wout16 = (unsigned short*)(ws + OFF_WOUT16);
    unsigned short* zx16   = (unsigned short*)(ws + OFF_ZX16);
    unsigned short* xbc16  = (unsigned short*)(ws + OFF_XBC16);
    float* dtc    = (float*)(ws + OFF_DTC);
    float* dAcs   = (float*)(ws + OFF_DACS);
    float* cdecay = (float*)(ws + OFF_CDEC);
    float* states = (float*)(ws + OFF_STATES);
    float* hprev  = (float*)(ws + OFF_HPREV);
    unsigned short* y16 = (unsigned short*)(ws + OFF_Y16);

    f2b_convert_kernel<<<1024, 256, 0, stream>>>(u, u16, (size_t)NTOK*D_MODEL);
    f2b_convert_kernel<<<1024, 256, 0, stream>>>(W_in, Win16, (size_t)D_IN_PROJ*D_MODEL);
    f2b_convert_kernel<<<1024, 256, 0, stream>>>(W_out, Wout16, (size_t)D_MODEL*D_INNER);

    dim3 g1(D_IN_PROJ/64, NTOK/128);
    gemm_xwt_kernel<false><<<g1, 256, 0, stream>>>(u16, Win16, zx16,
                                                   NTOK, D_IN_PROJ, D_MODEL);

    conv_silu_kernel<<<((size_t)NTOK*CONV_DIM + 255)/256, 256, 0, stream>>>(
        zx16, conv_w, conv_b, xbc16);

    dt_cumsum_kernel<<<BSZ*NC*NH, CHUNK, 0, stream>>>(zx16, dt_bias, A_log,
                                                      dtc, dAcs, cdecay);

    states_kernel<<<BSZ*NC*NH, 256, 0, stream>>>(xbc16, dtc, dAcs, states);

    hscan_kernel<<<BSZ*NH, 256, 0, stream>>>(states, cdecay, hprev);

    y_chunk_kernel<<<BSZ*NC*NH*4, 256, 0, stream>>>(xbc16, dtc, dAcs, hprev, Dp, y16);

    gatenorm_kernel<<<NTOK, 256, 0, stream>>>(zx16, norm_w, y16);

    dim3 g2(D_MODEL/64, NTOK/128);
    gemm_xwt_kernel<true><<<g2, 256, 0, stream>>>(y16, Wout16, d_out,
                                                  NTOK, D_MODEL, D_INNER);
}
